// DnnDependencyParser_69758858821741
// MI455X (gfx1250) — matrix operand order repaired and checked by reading, not compiled, not run on hardware
//
#include <hip/hip_runtime.h>

typedef __bf16 bf16;
typedef __attribute__((ext_vector_type(16))) __bf16 v16bf;
typedef __attribute__((ext_vector_type(8)))  float  v8f;
typedef unsigned int u32;
typedef u32 u32x4 __attribute__((ext_vector_type(4)));
typedef int i32x8 __attribute__((ext_vector_type(8)));
typedef int i32x4 __attribute__((ext_vector_type(4)));

#define T_SEQ 8192
#define HID   512
#define GATES 2048   // 4*H

#if defined(__has_builtin)
#if __has_builtin(__builtin_amdgcn_tensor_load_to_lds) && __has_builtin(__builtin_amdgcn_s_wait_tensorcnt)
#define USE_TDM 1
#endif
#endif
#ifndef USE_TDM
#define USE_TDM 0
#endif

// ---------------------------------------------------------------------------
// Embedding gather + concat + pad 105 -> 128, fp32 -> bf16
// ---------------------------------------------------------------------------
__global__ void pack_x_kernel(const int* __restrict__ widx, const int* __restrict__ pidx,
                              const float* __restrict__ wemb, const float* __restrict__ pemb,
                              bf16* __restrict__ xbf) {
  int t = blockIdx.x;
  int k = threadIdx.x;           // 0..127
  float v = 0.0f;
  if (k < 100)      v = wemb[(size_t)widx[t] * 100 + k];
  else if (k < 105) v = pemb[(size_t)pidx[t] * 5 + (k - 100)];
  xbf[(size_t)t * 128 + k] = (bf16)v;
}

// W_ih layer0: [2048,105] fp32 -> [2048,128] bf16 (zero-padded K)
__global__ void pad_wih0_kernel(const float* __restrict__ src, bf16* __restrict__ dst) {
  int r = blockIdx.x, k = threadIdx.x;
  float v = (k < 105) ? src[(size_t)r * 105 + k] : 0.0f;
  dst[(size_t)r * 128 + k] = (bf16)v;
}

__global__ void f2bf_kernel(const float* __restrict__ src, bf16* __restrict__ dst, size_t n) {
  size_t i = (size_t)blockIdx.x * blockDim.x + threadIdx.x;
  size_t stride = (size_t)gridDim.x * blockDim.x;
  for (; i < n; i += stride) dst[i] = (bf16)src[i];
}

// ---------------------------------------------------------------------------
// WMMA GEMM: out[T,2048](bf16) = A[T,K](bf16) x W[2048,K]^T(bf16) + bias(f32)
// Per-wave tile: 32(M) x 64(N) = 2 A-tiles x 4 B-tiles, K-loop step 32.
// B reused 2x, A reused 4x -> ~0.75 KB of L2 traffic per v_wmma (16 KFLOP).
// Lane layouts follow CDNA5 ISA 7.12.2.
// ---------------------------------------------------------------------------
__global__ __launch_bounds__(256) void gemm_wmma_kernel(
    const bf16* __restrict__ A, const bf16* __restrict__ W,
    const float* __restrict__ bias, bf16* __restrict__ out, int K) {
  const int N = GATES;
  const int wave = threadIdx.x >> 5;
  const int lane = threadIdx.x & 31;
  const int nstrips = N / 64;
  const int gt = blockIdx.x * 8 + wave;
  const int mt = gt / nstrips, ns = gt % nstrips;
  const int m0 = mt * 32, n0 = ns * 64;
  const int l16 = lane & 15;
  const int hi  = (lane >> 4) & 1;
  const int a0off = hi ? 8 : 0;          // lanes>=16 own K odd-8 chunks

  const bf16* arow0 = A + (size_t)(m0 + l16) * K;
  const bf16* arow1 = A + (size_t)(m0 + 16 + l16) * K;

  v8f acc[2][4] = {};
  union V16 { v16bf v; uint4 q[2]; };

  for (int k = 0; k < K; k += 32) {
    V16 a0, a1;
    a0.q[0] = *(const uint4*)(arow0 + k + a0off);
    a0.q[1] = *(const uint4*)(arow0 + k + 16 + a0off);
    a1.q[0] = *(const uint4*)(arow1 + k + a0off);
    a1.q[1] = *(const uint4*)(arow1 + k + 16 + a0off);
    const int koff = k + (hi ? 8 : 0);
#pragma unroll
    for (int nt = 0; nt < 4; ++nt) {
      const bf16* wrow = W + (size_t)(n0 + nt * 16 + l16) * K + koff;
      V16 b;
      b.q[0] = *(const uint4*)(wrow);
      b.q[1] = *(const uint4*)(wrow + 16);
      acc[0][nt] = __builtin_amdgcn_wmma_f32_16x16x32_bf16(
          false, a0.v, false, b.v, (short)0, acc[0][nt], false, false);
      acc[1][nt] = __builtin_amdgcn_wmma_f32_16x16x32_bf16(
          false, a1.v, false, b.v, (short)0, acc[1][nt], false, false);
    }
  }

#pragma unroll
  for (int mi = 0; mi < 2; ++mi) {
#pragma unroll
    for (int nt = 0; nt < 4; ++nt) {
      const int col = n0 + nt * 16 + l16;
      const float bv = bias[col];
#pragma unroll
      for (int r = 0; r < 8; ++r) {
        const int row = m0 + mi * 16 + (hi ? 8 : 0) + r;
        out[(size_t)row * N + col] = (bf16)(acc[mi][nt][r] + bv);
      }
    }
  }
}

// ---------------------------------------------------------------------------
// LSTM recurrence: 16 WGs per direction (grid=32, 1024 thr). Each WG owns 32
// h-dims -> 128 gate rows of W_hh staged into 128 KB of LDS via ONE Tensor
// Data Mover descriptor (4 gate blocks as a strided 2D tile: 16384 elem rows,
// stride 512*512). Per step: acquire-spin on L2 flag, 64-MAC dot + shfl
// reduce, 32 writer lanes apply i,f,g,o and release h chunk to L2.
// ---------------------------------------------------------------------------
__global__ __launch_bounds__(1024) void lstm_rec_kernel(
    const bf16* __restrict__ xW,    // [2][T][2048]  precomputed x@Wih^T+b
    const bf16* __restrict__ Whh,   // [2][2048][512]
    float* __restrict__ h_buf,      // [2][T][512]   step-ordered handoff
    unsigned* __restrict__ flags,   // [2][T]        completion counters
    bf16* __restrict__ h_out) {     // [T][1024]     fwd cols 0..511, bwd 512..1023
  extern __shared__ char smem_raw[];
  bf16*  slice = (bf16*)smem_raw;                        // [128][512]
  float* h_sm  = (float*)(smem_raw + 128 * HID * 2);     // [512]
  float* gates = h_sm + HID;                              // [128]
  float* c_sm  = gates + 128;                             // [32]

  const int dir = blockIdx.x >> 4;   // 0 fwd, 1 bwd
  const int wg  = blockIdx.x & 15;
  const int tid = threadIdx.x;
  const int row  = tid >> 3;         // 0..127 local gate row
  const int sub  = tid & 7;          // 8 lanes per row
  const int gate = row >> 5;         // i,f,g,o
  const int d    = row & 31;         // h-dim within this WG's chunk
  const int grow = gate * HID + wg * 32 + d;  // global gate row

#if USE_TDM
  if (tid < 32) {
    // Tensor DMA: 2D tile, 4 rows (gates) x 16384 bf16 (32 h-rows each),
    // tensor row stride = HID*HID elements, dest = contiguous LDS slice.
    const unsigned lds_off = (unsigned)(unsigned long long)(size_t)slice;
    const unsigned long long gaddr =
        (unsigned long long)(size_t)(Whh + ((size_t)dir * GATES + wg * 32) * HID);
    u32x4 g0;
    g0[0] = 1u;                                        // count=1, user descriptor
    g0[1] = lds_off;                                   // lds_addr
    g0[2] = (u32)gaddr;                                // global_addr[31:0]
    g0[3] = (u32)((gaddr >> 32) & 0x01FFFFFFu) | (2u << 30);  // addr[56:32] | type=2
    i32x8 g1;
    g1[0] = (int)(1u << 16);                           // data_size = 2 bytes
    g1[1] = (int)(16384u << 16);                       // tensor_dim0[15:0] = 16384
    g1[2] = (int)(4u << 16);                           // tensor_dim1 = 4
    g1[3] = (int)(16384u << 16);                       // tile_dim0 = 16384
    g1[4] = 4;                                         // tile_dim1 = 4
    g1[5] = (int)(HID * HID);                          // tensor_dim0_stride = 262144
    g1[6] = 0;
    g1[7] = 0;
    i32x4 gz4 = {0, 0, 0, 0};
    i32x8 gz8 = {0, 0, 0, 0, 0, 0, 0, 0};
    __builtin_amdgcn_tensor_load_to_lds(g0, g1, gz4, gz4, gz8, 0);
    __builtin_amdgcn_s_wait_tensorcnt(0);
  }
#else
  { // fallback: cooperative vector copy of the W_hh slice
    const bf16* src = Whh + ((size_t)dir * GATES + grow) * HID + sub * 64;
    bf16* dst = slice + (size_t)row * HID + sub * 64;
#pragma unroll
    for (int i = 0; i < 8; ++i) ((uint4*)dst)[i] = ((const uint4*)src)[i];
  }
#endif
  if (tid < 32) c_sm[tid] = 0.0f;
  __syncthreads();

  for (int s = 0; s < T_SEQ; ++s) {
    const int tt = (dir == 0) ? s : (T_SEQ - 1 - s);

    if (s > 0 && tid == 0) {
      while (__hip_atomic_load(&flags[(size_t)dir * T_SEQ + s - 1],
                               __ATOMIC_ACQUIRE, __HIP_MEMORY_SCOPE_AGENT) < 16u)
        __builtin_amdgcn_s_sleep(1);
    }
    __syncthreads();

    if (tid < HID) {
      float hv = 0.0f;
      if (s > 0)
        hv = __hip_atomic_load(&h_buf[((size_t)dir * T_SEQ + (s - 1)) * HID + tid],
                               __ATOMIC_RELAXED, __HIP_MEMORY_SCOPE_AGENT);
      h_sm[tid] = hv;
    }
    if (tid < 4 && s + 8 < T_SEQ) {  // prefetch xW gate chunks 8 steps ahead
      const int tp = (dir == 0) ? (s + 8) : (T_SEQ - 1 - (s + 8));
      __builtin_prefetch(&xW[((size_t)dir * T_SEQ + tp) * GATES + tid * HID + wg * 32], 0, 1);
    }
    __syncthreads();

    { // partial dot: 64 MACs per lane out of the 512-length row
      const bf16*  wrow = slice + (size_t)row * HID + sub * 64;
      const float* hseg = h_sm + sub * 64;
      float acc = 0.0f;
#pragma unroll
      for (int k = 0; k < 64; ++k) acc += (float)wrow[k] * hseg[k];
      acc += __shfl_xor(acc, 4, 8);
      acc += __shfl_xor(acc, 2, 8);
      acc += __shfl_xor(acc, 1, 8);
      if (sub == 0)
        gates[row] = acc + (float)xW[((size_t)dir * T_SEQ + tt) * GATES + grow];
    }
    __syncthreads();

    if (tid < 32) {
      const int dd = tid;
      float gi = gates[dd], gf = gates[32 + dd], gg = gates[64 + dd], go = gates[96 + dd];
      gi = 1.0f / (1.0f + __expf(-gi));
      gf = 1.0f / (1.0f + __expf(-gf));
      gg = tanhf(gg);
      go = 1.0f / (1.0f + __expf(-go));
      const float c = gf * c_sm[dd] + gi * gg;
      const float h = go * tanhf(c);
      c_sm[dd] = c;
      __hip_atomic_store(&h_buf[((size_t)dir * T_SEQ + s) * HID + wg * 32 + dd], h,
                         __ATOMIC_RELAXED, __HIP_MEMORY_SCOPE_AGENT);
      h_out[(size_t)tt * 1024 + dir * HID + wg * 32 + dd] = (bf16)h;
    }
    __threadfence();
    __syncthreads();
    if (tid == 0)
      __hip_atomic_fetch_add(&flags[(size_t)dir * T_SEQ + s], 1u,
                             __ATOMIC_RELEASE, __HIP_MEMORY_SCOPE_AGENT);
  }
}

// ---------------------------------------------------------------------------
// Per-token head / modifier scores
// ---------------------------------------------------------------------------
__global__ void proj_kernel(const bf16* __restrict__ v,
                            const float* __restrict__ Wh, const float* __restrict__ bh,
                            const float* __restrict__ Wm, const float* __restrict__ bm,
                            float* __restrict__ heads, float* __restrict__ mods) {
  int t = blockIdx.x * blockDim.x + threadIdx.x;
  if (t >= T_SEQ) return;
  float sh = 0.0f, sm = 0.0f;
  const bf16* vr = v + (size_t)t * 1024;
  for (int k = 0; k < 1024; ++k) {
    const float x = (float)vr[k];
    sh += x * Wh[k];
    sm += x * Wm[k];
  }
  heads[t] = sh + bh[0];
  mods[t]  = sm + bm[0];
}

// ---------------------------------------------------------------------------
// table[i,j] = heads[i] + mods[j]  (256 MB of float4 stores -> HBM-bound)
// ---------------------------------------------------------------------------
__global__ void table_kernel(const float* __restrict__ heads,
                             const float* __restrict__ mods,
                             float* __restrict__ out) {
  const size_t total4 = (size_t)T_SEQ * T_SEQ / 4;
  size_t idx = (size_t)blockIdx.x * blockDim.x + threadIdx.x;
  const size_t stride = (size_t)gridDim.x * blockDim.x;
  for (; idx < total4; idx += stride) {
    const size_t i = idx / (T_SEQ / 4);
    const size_t j = (idx % (T_SEQ / 4)) * 4;
    const float h = heads[i];
    const float4 m = *(const float4*)(mods + j);
    float4 r; r.x = h + m.x; r.y = h + m.y; r.z = h + m.z; r.w = h + m.w;
    *(float4*)(out + i * (size_t)T_SEQ + j) = r;
  }
}

// ---------------------------------------------------------------------------
extern "C" void kernel_launch(void* const* d_in, const int* in_sizes, int n_in,
                              void* d_out, int out_size, void* d_ws, size_t ws_size,
                              hipStream_t stream) {
  (void)in_sizes; (void)n_in; (void)out_size; (void)ws_size;
  const int T = T_SEQ;
  const int* widx = (const int*)d_in[0];
  const int* pidx = (const int*)d_in[1];
  // d_in[2] = head_tensor: unused by the reference output
  const float* wemb = (const float*)d_in[3];
  const float* pemb = (const float*)d_in[4];
  const float* Wih_l0  = (const float*)d_in[5];
  const float* Whh_l0  = (const float*)d_in[6];
  const float* b_l0    = (const float*)d_in[7];
  const float* Wih_l0r = (const float*)d_in[8];
  const float* Whh_l0r = (const float*)d_in[9];
  const float* b_l0r   = (const float*)d_in[10];
  const float* Wih_l1  = (const float*)d_in[11];
  const float* Whh_l1  = (const float*)d_in[12];
  const float* b_l1    = (const float*)d_in[13];
  const float* Wih_l1r = (const float*)d_in[14];
  const float* Whh_l1r = (const float*)d_in[15];
  const float* b_l1r   = (const float*)d_in[16];
  const float* W_head = (const float*)d_in[17];
  const float* b_head = (const float*)d_in[18];
  const float* W_mod  = (const float*)d_in[19];
  const float* b_mod  = (const float*)d_in[20];

  char* w = (char*)d_ws;
  auto alloc = [&](size_t bytes) { char* p = w; w += (bytes + 255) & ~(size_t)255; return p; };
  bf16*  x_bf  = (bf16*)alloc((size_t)T * 128 * 2);                 //   2 MB
  bf16*  wih0b = (bf16*)alloc((size_t)2 * GATES * 128 * 2);         //   1 MB
  bf16*  whh0b = (bf16*)alloc((size_t)2 * GATES * HID * 2);         //   4 MB
  bf16*  whh1b = (bf16*)alloc((size_t)2 * GATES * HID * 2);         //   4 MB
  bf16*  wih1b = (bf16*)alloc((size_t)2 * GATES * 1024 * 2);        //   8 MB
  bf16*  xW    = (bf16*)alloc((size_t)2 * T * GATES * 2);           //  64 MB (reused l0/l1)
  float* h_buf = (float*)alloc((size_t)2 * T * HID * 4);            //  32 MB (reused)
  bf16*  h_l0  = (bf16*)alloc((size_t)T * 1024 * 2);                //  16 MB
  bf16*  vbuf  = (bf16*)alloc((size_t)T * 1024 * 2);                //  16 MB
  float* heads = (float*)alloc((size_t)T * 4);
  float* mods  = (float*)alloc((size_t)T * 4);
  unsigned* flags = (unsigned*)alloc((size_t)2 * 2 * T * 4);        // [layer][dir][T]

  (void)hipMemsetAsync(flags, 0, (size_t)2 * 2 * T * 4, stream);

  pack_x_kernel<<<T, 128, 0, stream>>>(widx, pidx, wemb, pemb, x_bf);
  pad_wih0_kernel<<<GATES, 128, 0, stream>>>(Wih_l0,  wih0b);
  pad_wih0_kernel<<<GATES, 128, 0, stream>>>(Wih_l0r, wih0b + (size_t)GATES * 128);
  f2bf_kernel<<<512, 256, 0, stream>>>(Whh_l0,  whh0b,                       (size_t)GATES * HID);
  f2bf_kernel<<<512, 256, 0, stream>>>(Whh_l0r, whh0b + (size_t)GATES * HID, (size_t)GATES * HID);
  f2bf_kernel<<<512, 256, 0, stream>>>(Whh_l1,  whh1b,                       (size_t)GATES * HID);
  f2bf_kernel<<<512, 256, 0, stream>>>(Whh_l1r, whh1b + (size_t)GATES * HID, (size_t)GATES * HID);
  f2bf_kernel<<<1024, 256, 0, stream>>>(Wih_l1,  wih1b,                        (size_t)GATES * 1024);
  f2bf_kernel<<<1024, 256, 0, stream>>>(Wih_l1r, wih1b + (size_t)GATES * 1024, (size_t)GATES * 1024);

  const int gemm_blocks = (T / 32) * (GATES / 64) / 8;   // 1024 blocks, 8 waves each
  const size_t rec_lds = (size_t)128 * HID * 2 + HID * 4 + 128 * 4 + 32 * 4;

  // Layer 0: input projection (K=128) then recurrence
  gemm_wmma_kernel<<<gemm_blocks, 256, 0, stream>>>(x_bf, wih0b, b_l0, xW, 128);
  gemm_wmma_kernel<<<gemm_blocks, 256, 0, stream>>>(x_bf, wih0b + (size_t)GATES * 128,
                                                    b_l0r, xW + (size_t)T * GATES, 128);
  lstm_rec_kernel<<<32, 1024, rec_lds, stream>>>(xW, whh0b, h_buf, flags, h_l0);

  // Layer 1: input projection (K=1024) then recurrence
  gemm_wmma_kernel<<<gemm_blocks, 256, 0, stream>>>(h_l0, wih1b, b_l1, xW, 1024);
  gemm_wmma_kernel<<<gemm_blocks, 256, 0, stream>>>(h_l0, wih1b + (size_t)GATES * 1024,
                                                    b_l1r, xW + (size_t)T * GATES, 1024);
  lstm_rec_kernel<<<32, 1024, rec_lds, stream>>>(xW, whh1b, h_buf, flags + (size_t)2 * T, vbuf);

  proj_kernel<<<T / 256, 256, 0, stream>>>(vbuf, W_head, b_head, W_mod, b_mod, heads, mods);
  table_kernel<<<4096, 256, 0, stream>>>(heads, mods, (float*)d_out);
}
